// CAKT_57578331570315
// MI455X (gfx1250) — matrix operand order repaired and checked by reading, not compiled, not run on hardware
//
#include <hip/hip_runtime.h>
#include <hip/hip_bf16.h>
#include <math.h>

// ---------------------------------------------------------------------------
// CDNA5 / gfx1250 fused AKT attention.  wave32, WMMA bf16 (16x16x32),
// TDM tensor_load_to_lds + async global->LDS staging, contiguous fragment
// reads (ds_load_b128 pairs).
// ---------------------------------------------------------------------------

typedef __attribute__((ext_vector_type(16))) __bf16 v16bf;
typedef __attribute__((ext_vector_type(8)))  float  v8f;
typedef __attribute__((ext_vector_type(4)))  unsigned int u32x4;
typedef __attribute__((ext_vector_type(4)))  int  i32x4;
typedef __attribute__((ext_vector_type(8)))  int  i32x8;

union BfFrag { v16bf v; unsigned short u[16]; };

__device__ __forceinline__ unsigned short f2bf(float f) {
  unsigned int x = __builtin_bit_cast(unsigned int, f);
  x += 0x7FFFu + ((x >> 16) & 1u);          // round-to-nearest-even
  return (unsigned short)(x >> 16);
}

#define WMMA_BF16(A, B, C) \
  __builtin_amdgcn_wmma_f32_16x16x32_bf16(false, (A), false, (B), (short)0, (C), false, false)

// ---- gfx1250 async global->LDS copy (ASYNCcnt-tracked) --------------------
__device__ __forceinline__ void async_ldg_b128(unsigned lds_off, const void* g) {
  unsigned long long ga = (unsigned long long)g;
  asm volatile("global_load_async_to_lds_b128 %0, %1, off"
               :: "v"(lds_off), "v"(ga) : "memory");
}
__device__ __forceinline__ void wait_async0() {
#if __has_builtin(__builtin_amdgcn_s_wait_asynccnt)
  __builtin_amdgcn_s_wait_asynccnt(0);
#else
  asm volatile("s_wait_asynccnt 0" ::: "memory");
#endif
}

// ---- gfx1250 Tensor Data Mover: DMA a 64x64 bf16 tile into padded LDS -----
// D# per ISA 8.3/8.4: count=1, type=2, data_size=2B, pad 4 DWORDs after each
// 32 DWORDs (row stride 64 halfs + 8 pad = 144B), tensor 512x64, tile 64x64.
#if __has_builtin(__builtin_amdgcn_tensor_load_to_lds) && \
    __has_builtin(__builtin_amdgcn_s_wait_tensorcnt)
#define HAVE_TDM 1
#else
#define HAVE_TDM 0
#endif

#if HAVE_TDM
__device__ __forceinline__ void tdm_load_tile_64x64_bf16(unsigned lds_byte_addr,
                                                         const void* gsrc) {
  const unsigned long long ga = (unsigned long long)gsrc;
  u32x4 g0;
  g0.x = 1u;                                                // count=1 (valid)
  g0.y = lds_byte_addr;                                     // lds_addr
  g0.z = (unsigned)ga;                                      // global_addr[31:0]
  g0.w = (unsigned)((ga >> 32) & 0x1FFFFFFu) | (2u << 30);  // [56:32] | type=2
  i32x8 g1;
  g1[0] = (int)((1u << 16)     // data_size = 2 bytes
              | (1u << 20)     // pad_enable
              | (4u << 22)     // pad_interval: 32 DWORDs (=128B row)
              | (3u << 25));   // pad_amount:   4 DWORDs (=16B  pad)
  g1[1] = (int)(64u << 16);    // tensor_dim0 = 64 (d axis)
  g1[2] = (int)(512u << 16);   // tensor_dim1 = 512 (row axis)
  g1[3] = (int)(64u << 16);    // tile_dim0 = 64
  g1[4] = 64;                  // tile_dim1 = 64, tile_dim2 = 0
  g1[5] = 64;                  // tensor_dim0_stride = 64
  g1[6] = 0; g1[7] = 0;
  const i32x4 z4 = {0, 0, 0, 0};
#if defined(__clang_major__) && (__clang_major__ >= 23)
  const i32x8 z8 = {0, 0, 0, 0, 0, 0, 0, 0};
  __builtin_amdgcn_tensor_load_to_lds(g0, g1, z4, z4, z8, 0);
#else
  __builtin_amdgcn_tensor_load_to_lds(g0, g1, z4, z4, 0);
#endif
}
#endif

// ---------------------------------------------------------------------------
// GEMM:  out_bf16[head-split] = X_f32[M,512] @ W_f32[512,512]^T + bias
// 256 thr (8 waves); 64x64 tile; wave w -> (mi=w&3, ni={2*(w>>2),+1}).
// Staging: batch all 8 global b128 loads, then convert+store (one wait).
// ---------------------------------------------------------------------------
__global__ __launch_bounds__(256) void akt_proj_gemm(
    const float* __restrict__ X, const float* __restrict__ W,
    const float* __restrict__ bias, unsigned short* __restrict__ out)
{
  __shared__ unsigned short At[64][72];   // A[m][k] bf16
  __shared__ unsigned short Wl[64][72];   // W[n][k] bf16 (natural)
  const int m0 = blockIdx.x * 64, n0 = blockIdx.y * 64;
  const int tid = threadIdx.x, lane = tid & 31, wv = tid >> 5;
  const int mi = wv & 3, nb = (wv >> 2) << 1;
  v8f acc0 = {}, acc1 = {};

  for (int k0 = 0; k0 < 512; k0 += 64) {
    const int r = tid >> 4, c4 = (tid & 15) << 2;
    float4 fx[4], fw[4];
#pragma unroll
    for (int rr = 0; rr < 4; ++rr) {       // issue all loads first
      const int row = r + rr * 16;
      fx[rr] = *(const float4*)(X + (size_t)(m0 + row) * 512 + k0 + c4);
      fw[rr] = *(const float4*)(W + (size_t)(n0 + row) * 512 + k0 + c4);
    }
    if (k0 + 64 < 512) {                   // global_prefetch_b8 of next K tile
      __builtin_prefetch(X + (size_t)(m0 + r) * 512 + k0 + 64 + c4, 0, 1);
      __builtin_prefetch(W + (size_t)(n0 + r) * 512 + k0 + 64 + c4, 0, 1);
    }
#pragma unroll
    for (int rr = 0; rr < 4; ++rr) {       // convert + store
      const int row = r + rr * 16;
      At[row][c4 + 0] = f2bf(fx[rr].x); At[row][c4 + 1] = f2bf(fx[rr].y);
      At[row][c4 + 2] = f2bf(fx[rr].z); At[row][c4 + 3] = f2bf(fx[rr].w);
      Wl[row][c4 + 0] = f2bf(fw[rr].x); Wl[row][c4 + 1] = f2bf(fw[rr].y);
      Wl[row][c4 + 2] = f2bf(fw[rr].z); Wl[row][c4 + 3] = f2bf(fw[rr].w);
    }
    __syncthreads();
#pragma unroll
    for (int kk = 0; kk < 64; kk += 32) {
      BfFrag a, b0, b1;
      const int arow = mi * 16 + (lane & 15);
      const int kA = kk + ((lane >> 4) << 4);
      const int kB = kk + ((lane >> 4) << 4);
      const int n0l = nb * 16 + (lane & 15), n1l = n0l + 16;
#pragma unroll
      for (int i = 0; i < 16; ++i) {
        a.u[i]  = At[arow][kA + i];                    // 32B contiguous
        b0.u[i] = Wl[n0l][kB + i];                     // 32B contiguous
        b1.u[i] = Wl[n1l][kB + i];
      }
      acc0 = WMMA_BF16(a.v, b0.v, acc0);
      acc1 = WMMA_BF16(a.v, b1.v, acc1);
    }
    __syncthreads();
  }
  const int nl0 = n0 + nb * 16 + (lane & 15), nl1 = nl0 + 16;
  const float bia0 = bias[nl0], bia1 = bias[nl1];
  const int h0 = nl0 >> 6, dk0 = nl0 & 63;
  const int h1 = nl1 >> 6, dk1 = nl1 & 63;
#pragma unroll
  for (int v = 0; v < 8; ++v) {
    const int m = m0 + mi * 16 + v + ((lane >> 4) << 3);
    const int bb = m >> 9, s = m & 511;
    out[(size_t)(bb * 8 + h0) * 32768 + s * 64 + dk0] = f2bf(acc0[v] + bia0);
    out[(size_t)(bb * 8 + h1) * 32768 + s * 64 + dk1] = f2bf(acc1[v] + bia1);
  }
}

// ---------------------------------------------------------------------------
// GEMM:  out_f32 = X_bf16[M,512] @ W_f32[512,512]^T + bias  (final Wo proj)
// ---------------------------------------------------------------------------
__global__ __launch_bounds__(256) void akt_out_gemm(
    const unsigned short* __restrict__ Xb, const float* __restrict__ W,
    const float* __restrict__ bias, float* __restrict__ out)
{
  __shared__ unsigned short At[64][72];
  __shared__ unsigned short Wl[64][72];
  const int m0 = blockIdx.x * 64, n0 = blockIdx.y * 64;
  const int tid = threadIdx.x, lane = tid & 31, wv = tid >> 5;
  const int mi = wv & 3, nb = (wv >> 2) << 1;
  v8f acc0 = {}, acc1 = {};

  for (int k0 = 0; k0 < 512; k0 += 64) {
    const int r = tid >> 4, c4 = (tid & 15) << 2;
    ushort4 ux[4]; float4 fw[4];
#pragma unroll
    for (int rr = 0; rr < 4; ++rr) {
      const int row = r + rr * 16;
      ux[rr] = *(const ushort4*)(Xb + (size_t)(m0 + row) * 512 + k0 + c4);
      fw[rr] = *(const float4*)(W + (size_t)(n0 + row) * 512 + k0 + c4);
    }
#pragma unroll
    for (int rr = 0; rr < 4; ++rr) {
      const int row = r + rr * 16;
      At[row][c4 + 0] = ux[rr].x; At[row][c4 + 1] = ux[rr].y;
      At[row][c4 + 2] = ux[rr].z; At[row][c4 + 3] = ux[rr].w;
      Wl[row][c4 + 0] = f2bf(fw[rr].x); Wl[row][c4 + 1] = f2bf(fw[rr].y);
      Wl[row][c4 + 2] = f2bf(fw[rr].z); Wl[row][c4 + 3] = f2bf(fw[rr].w);
    }
    __syncthreads();
#pragma unroll
    for (int kk = 0; kk < 64; kk += 32) {
      BfFrag a, b0, b1;
      const int arow = mi * 16 + (lane & 15);
      const int kA = kk + ((lane >> 4) << 4);
      const int kB = kk + ((lane >> 4) << 4);
      const int n0l = nb * 16 + (lane & 15), n1l = n0l + 16;
#pragma unroll
      for (int i = 0; i < 16; ++i) {
        a.u[i]  = At[arow][kA + i];
        b0.u[i] = Wl[n0l][kB + i];
        b1.u[i] = Wl[n1l][kB + i];
      }
      acc0 = WMMA_BF16(a.v, b0.v, acc0);
      acc1 = WMMA_BF16(a.v, b1.v, acc1);
    }
    __syncthreads();
  }
  const int nl0 = n0 + nb * 16 + (lane & 15), nl1 = nl0 + 16;
  const float bia0 = bias[nl0], bia1 = bias[nl1];
#pragma unroll
  for (int v = 0; v < 8; ++v) {
    const int m = m0 + mi * 16 + v + ((lane >> 4) << 3);
    out[(size_t)m * 512 + nl0] = acc0[v] + bia0;
    out[(size_t)m * 512 + nl1] = acc1[v] + bia1;
  }
}

// ---------------------------------------------------------------------------
// Fused attention: one block per (q-block of 64, head, batch).
// LDS: S f32 strip [64][512] | Pb bf16 strip [64][512] (aliases Qt) | KV | red
// 202 KB dynamic LDS -- only possible with CDNA5's 320 KB/WGP.
// ---------------------------------------------------------------------------
#define OFF_PB   131072                     // also Qt ([64][72]) during scores
#define OFF_KV   (131072 + 65536)
#define OFF_RED  (131072 + 65536 + 9216)
#define ATTN_LDS_BYTES (131072 + 65536 + 9216 + 1024)

__global__ __launch_bounds__(256) void akt_attn(
    const unsigned short* __restrict__ Qh,
    const unsigned short* __restrict__ Kh,
    const unsigned short* __restrict__ Vh,
    const float* __restrict__ gammas,
    const int*  __restrict__ zero_pad,
    unsigned short* __restrict__ concat)
{
  extern __shared__ char smem[];
  float*          S   = (float*)smem;                        // [64][512] f32
  unsigned short* Pb  = (unsigned short*)(smem + OFF_PB);    // [64][512] bf16 probs
  unsigned short* Qt  = (unsigned short*)(smem + OFF_PB);    // [64][72] (scores phase)
  unsigned short* KV  = (unsigned short*)(smem + OFF_KV);    // [64][72] bf16
  float*          red = (float*)(smem + OFF_RED);            // [64][4]

  const int b = blockIdx.z, h = blockIdx.y;
  const int q0 = blockIdx.x * 64;
  const size_t hb = (size_t)(b * 8 + h) * 32768;             // S*DK
  const unsigned short* Qp = Qh + hb;
  const unsigned short* Kp = Kh + hb;
  const unsigned short* Vp = Vh + hb;

  const int tid = threadIdx.x, lane = tid & 31, wv = tid >> 5;
  const int mi = wv & 3, nb = (wv >> 2) << 1;
  const unsigned ldsbase = __builtin_amdgcn_groupstaticsize();

  // ---- async-stage Q tile (natural [q][d]) into Qt ----------------------
#pragma unroll
  for (int j = 0; j < 2; ++j) {
    const int c = tid + 256 * j;                 // 512 chunks of 8 halfs
    const int row = c >> 3, col = (c & 7) * 8;
    async_ldg_b128(ldsbase + OFF_PB + (unsigned)(row * 72 + col) * 2,
                   Qp + (size_t)(q0 + row) * 64 + col);
  }

  // ---- scores = Q K^T * (1/sqrt(64)) -> LDS f32 strip -------------------
  for (int kb = 0; kb < 8; ++kb) {
    __syncthreads();                             // readers of prev tile done
#if HAVE_TDM
    if (wv == 0) {                               // one wave drives the TDM DMA
      tdm_load_tile_64x64_bf16(ldsbase + OFF_KV, Kp + (size_t)kb * 64 * 64);
      __builtin_amdgcn_s_wait_tensorcnt(0);
    }
#else
#pragma unroll
    for (int j = 0; j < 2; ++j) {                // async-stage K block kb
      const int c = tid + 256 * j;
      const int row = c >> 3, col = (c & 7) * 8;
      async_ldg_b128(ldsbase + OFF_KV + (unsigned)(row * 72 + col) * 2,
                     Kp + (size_t)(kb * 64 + row) * 64 + col);
    }
#endif
    wait_async0();                               // Q tile (1st iter) + fallback K
    __syncthreads();                             // tiles visible to all waves
    v8f acc0 = {}, acc1 = {};
#pragma unroll
    for (int kk = 0; kk < 64; kk += 32) {
      BfFrag a, b0, b1;
      const int arow = mi * 16 + (lane & 15);
      const int kA = kk + ((lane >> 4) << 4);
      const int kB = kk + ((lane >> 4) << 4);
      const int n0l = nb * 16 + (lane & 15), n1l = n0l + 16;
#pragma unroll
      for (int i = 0; i < 16; ++i) {
        a.u[i]  = Qt[arow * 72 + kA + i];
        b0.u[i] = KV[n0l * 72 + kB + i];   // B[d][key]=K[key][d]: rows are K-contig
        b1.u[i] = KV[n1l * 72 + kB + i];
      }
      acc0 = WMMA_BF16(a.v, b0.v, acc0);
      acc1 = WMMA_BF16(a.v, b1.v, acc1);
    }
    const int col0 = kb * 64 + nb * 16 + (lane & 15);
#pragma unroll
    for (int v = 0; v < 8; ++v) {
      const int r = mi * 16 + v + ((lane >> 4) << 3);
      S[r * 512 + col0]      = acc0[v] * 0.125f;
      S[r * 512 + col0 + 16] = acc1[v] * 0.125f;
    }
  }
  __syncthreads();

  // ---- softmax1 -> cumsum -> distance decay -> softmax2 (4 thr / row) ---
  const float g   = gammas[h];
  const float gam = -(g > 20.f ? g : log1pf(__expf(g)));   // -softplus(gamma)
  const int   zp  = zero_pad[0];

  const int r = tid >> 2, t = tid & 3;
  const int qg = q0 + r;                       // global query index
  float* Srow = S + r * 512;
  const int c_lo = t << 7;                     // 128-col segment per thread
  const int nallow = qg;                       // strict causal: cols [0,qg)
  const int c_end = (nallow < c_lo + 128) ? (nallow > c_lo ? nallow : c_lo) : (c_lo + 128);

  float mx = -3.0e38f;
  for (int c = c_lo; c < c_end; ++c) mx = fmaxf(mx, Srow[c]);
  red[r * 4 + t] = mx;
  __syncthreads();
  const float rmax = fmaxf(fmaxf(red[r*4+0], red[r*4+1]), fmaxf(red[r*4+2], red[r*4+3]));
  __syncthreads();
  float sm = 0.f;
  for (int c = c_lo; c < c_end; ++c) sm += __expf(Srow[c] - rmax);
  red[r * 4 + t] = sm;
  __syncthreads();
  const float rsum = red[r*4+0] + red[r*4+1] + red[r*4+2] + red[r*4+3];
  const float inv  = rsum > 0.f ? 1.f / rsum : 0.f;
  float cum = 0.f;
  for (int tt = 0; tt < t; ++tt) cum += red[r * 4 + tt];
  cum *= inv;
  const float total = nallow > 0 ? 1.f : 0.f;  // disttotal (softmax mass on allowed)
  __syncthreads();
  for (int c = c_lo; c < c_lo + 128; ++c) {
    const float raw = Srow[c];
    const float p = (c < nallow) ? __expf(raw - rmax) * inv : 0.f;
    cum += p;                                  // inclusive cumsum of scores_
    float ds = (total - cum) * fabsf((float)(qg - c));
    ds = ds > 0.f ? sqrtf(ds) : 0.f;
    float eff = __expf(ds * gam);
    eff = fminf(fmaxf(eff, 1e-5f), 1e5f);
    Srow[c] = raw * eff;
  }
  // second softmax over rescaled scores
  float mx2 = -3.0e38f;
  for (int c = c_lo; c < c_end; ++c) mx2 = fmaxf(mx2, Srow[c]);
  red[r * 4 + t] = mx2;
  __syncthreads();
  const float rmax2 = fmaxf(fmaxf(red[r*4+0], red[r*4+1]), fmaxf(red[r*4+2], red[r*4+3]));
  __syncthreads();
  float sm2 = 0.f;
  for (int c = c_lo; c < c_end; ++c) sm2 += __expf(Srow[c] - rmax2);
  red[r * 4 + t] = sm2;
  __syncthreads();
  const float rsum2 = red[r*4+0] + red[r*4+1] + red[r*4+2] + red[r*4+3];
  const float inv2  = rsum2 > 0.f ? 1.f / rsum2 : 0.f;
  for (int c = c_lo; c < c_lo + 128; ++c) {
    float pv;
    if (zp != 0 && qg == 0)   pv = 0.f;              // zero_pad: kill query row 0
    else if (nallow == 0)     pv = 1.f / 512.f;      // all-masked softmax is uniform
    else if (c < nallow)      pv = __expf(Srow[c] - rmax2) * inv2;
    else                      pv = 0.f;
    Pb[r * 512 + c] = f2bf(pv);                      // bf16 A operand for P@V
  }

  // ---- out = P @ V ------------------------------------------------------
  v8f o0 = {}, o1 = {};
  for (int kb = 0; kb < 8; ++kb) {
    __syncthreads();
    // stage V block TRANSPOSED: Vt[d][key] so B fragments are K-contiguous
    for (int i = tid; i < 1024; i += 256) {
      const int key = i >> 4, d4 = (i & 15) << 2;
      const ushort4 vv = *(const ushort4*)(Vp + (size_t)(kb * 64 + key) * 64 + d4);
      KV[(d4 + 0) * 72 + key] = vv.x; KV[(d4 + 1) * 72 + key] = vv.y;
      KV[(d4 + 2) * 72 + key] = vv.z; KV[(d4 + 3) * 72 + key] = vv.w;
    }
    __syncthreads();
#pragma unroll
    for (int kk = 0; kk < 64; kk += 32) {
      BfFrag a, b0, b1;
      const int arow = mi * 16 + (lane & 15);
      const int kA = kb * 64 + kk + ((lane >> 4) << 4);
      const int kB = kk + ((lane >> 4) << 4);
      const int n0l = nb * 16 + (lane & 15), n1l = n0l + 16;
#pragma unroll
      for (int i = 0; i < 16; ++i) {
        a.u[i]  = Pb[arow * 512 + kA + i];                    // contiguous bf16
        b0.u[i] = KV[n0l * 72 + kB + i];                      // Vt rows K-contig
        b1.u[i] = KV[n1l * 72 + kB + i];
      }
      o0 = WMMA_BF16(a.v, b0.v, o0);
      o1 = WMMA_BF16(a.v, b1.v, o1);
    }
  }
  const int d0 = nb * 16 + (lane & 15);
#pragma unroll
  for (int v = 0; v < 8; ++v) {
    const int rr = mi * 16 + v + ((lane >> 4) << 3);
    const size_t base = ((size_t)(b * 512 + q0 + rr)) * 512 + h * 64;
    concat[base + d0]      = f2bf(o0[v]);
    concat[base + d0 + 16] = f2bf(o1[v]);
  }
}

// ---------------------------------------------------------------------------
extern "C" void kernel_launch(void* const* d_in, const int* in_sizes, int n_in,
                              void* d_out, int out_size, void* d_ws, size_t ws_size,
                              hipStream_t stream) {
  (void)in_sizes; (void)n_in; (void)out_size;
  const float* q       = (const float*)d_in[0];
  const float* k       = (const float*)d_in[1];
  const float* v       = (const float*)d_in[2];
  // d_in[3] = mask (recomputed analytically, unused)
  const int*   zero_pad= (const int*)d_in[4];
  const float* Wq      = (const float*)d_in[5];
  const float* bq      = (const float*)d_in[6];
  const float* Wk      = (const float*)d_in[7];
  const float* bk      = (const float*)d_in[8];
  const float* Wv      = (const float*)d_in[9];
  const float* bv      = (const float*)d_in[10];
  const float* Wo      = (const float*)d_in[11];
  const float* bo      = (const float*)d_in[12];
  const float* gammas  = (const float*)d_in[13];

  const size_t HEAD_BYTES = (size_t)32 * 8 * 512 * 64 * 2;  // 16 MiB each, bf16
  if (ws_size < 4 * HEAD_BYTES) return;
  unsigned short* Qh     = (unsigned short*)d_ws;
  unsigned short* Kh     = (unsigned short*)((char*)d_ws + HEAD_BYTES);
  unsigned short* Vh     = (unsigned short*)((char*)d_ws + 2 * HEAD_BYTES);
  unsigned short* concat = (unsigned short*)((char*)d_ws + 3 * HEAD_BYTES);

  dim3 blk(256);
  dim3 gG(16384 / 64, 512 / 64);          // 256 x 8 tiles
  akt_proj_gemm<<<gG, blk, 0, stream>>>(q, Wq, bq, Qh);
  akt_proj_gemm<<<gG, blk, 0, stream>>>(k, Wk, bk, Kh);
  akt_proj_gemm<<<gG, blk, 0, stream>>>(v, Wv, bv, Vh);

  hipFuncSetAttribute(reinterpret_cast<const void*>(akt_attn),
                      hipFuncAttributeMaxDynamicSharedMemorySize, ATTN_LDS_BYTES);
  dim3 gA(8, 8, 32);                      // (q-blocks, heads, batch)
  akt_attn<<<gA, blk, ATTN_LDS_BYTES, stream>>>(Qh, Kh, Vh, gammas, zero_pad, concat);

  akt_out_gemm<<<gG, blk, 0, stream>>>(concat, Wo, bo, (float*)d_out);
}
